// MHAAttention2D_3075196584136
// MI455X (gfx1250) — matrix operand order repaired and checked by reading, not compiled, not run on hardware
//
#include <hip/hip_runtime.h>

// ---------- types ----------
typedef __attribute__((ext_vector_type(16))) __bf16 v16bf;
typedef __attribute__((ext_vector_type(8)))  __bf16 v8bf;
typedef __attribute__((ext_vector_type(8)))  float  v8f;
typedef __attribute__((ext_vector_type(4)))  int    v4i;

#define WMMA_BF16(a, b, c) \
  __builtin_amdgcn_wmma_f32_16x16x32_bf16(false, (a), false, (b), (short)0, (c), false, false)

#if __has_builtin(__builtin_amdgcn_global_load_async_to_lds_b128)
#define HAS_ASYNC_LDS 1
#else
#define HAS_ASYNC_LDS 0
#endif

#define AS1 __attribute__((address_space(1)))
#define AS3 __attribute__((address_space(3)))

__device__ inline void async_copy_b128(const __bf16* g, __bf16* l) {
#if HAS_ASYNC_LDS
  __builtin_amdgcn_global_load_async_to_lds_b128(
      (AS1 v4i*)(const AS1 void*)g, (AS3 v4i*)(AS3 void*)l, 0, 0);
#else
  *reinterpret_cast<v8bf*>(l) = *reinterpret_cast<const v8bf*>(g);
#endif
}

__device__ inline void wait_async_copies() {
#if HAS_ASYNC_LDS
#if __has_builtin(__builtin_amdgcn_s_wait_asynccnt)
  __builtin_amdgcn_s_wait_asynccnt(0);
#else
  asm volatile("s_wait_asynccnt 0x0" ::: "memory");
#endif
#endif
}

// A-matrix fragment (16x32 bf16, M x K), base = row-major [16 x ld], rows = M.
// ISA 7.12.2: lanes 0-15 -> M=lane, VGPR0..3 = K0..7, VGPR4..7 = K16..23
//             lanes16-31 -> M=lane-16, VGPR0..3 = K8..15, VGPR4..7 = K24..31
__device__ inline v16bf load_frag_a(const __bf16* base, int ld) {
  int lane = threadIdx.x & 31;
  int m = lane & 15, half = lane >> 4;
  const __bf16* row = base + m * ld;
  v8bf lo = *reinterpret_cast<const v8bf*>(row + half * 8);
  v8bf hi = *reinterpret_cast<const v8bf*>(row + 16 + half * 8);
  v16bf f;
#pragma unroll
  for (int i = 0; i < 8; ++i) { f[i] = lo[i]; f[i + 8] = hi[i]; }
  return f;
}

// B-matrix fragment (32x16 bf16, K x N), base = row-major B^T = [16(N) x ld(K...)].
// lanes 0-15 (n=lane) hold K=0..7 in VGPR0..3, K=16..23 in VGPR4..7; lanes 16-31 hold K=8..15, K=24..31.
__device__ inline v16bf load_frag_b(const __bf16* base, int ld) {
  int lane = threadIdx.x & 31;
  int n = lane & 15, half = lane >> 4;
  const __bf16* row = base + n * ld;
  v8bf lo = *reinterpret_cast<const v8bf*>(row + half * 8);
  v8bf hi = *reinterpret_cast<const v8bf*>(row + 16 + half * 8);
  v16bf f;
#pragma unroll
  for (int i = 0; i < 8; ++i) { f[i] = lo[i]; f[i + 8] = hi[i]; }
  return f;
}

// ---------- constants ----------
#define CDIM 512
#define NTOK 4096
#define HEADS 16
#define HDIM 32

// ---------- weight convert fp32 -> bf16 ----------
__global__ void cvt_w_kernel(const float* __restrict__ wq, const float* __restrict__ wk,
                             const float* __restrict__ wv, const float* __restrict__ wo,
                             __bf16* __restrict__ oq, __bf16* __restrict__ ok,
                             __bf16* __restrict__ ov, __bf16* __restrict__ oo) {
  int i = blockIdx.x * 256 + threadIdx.x;
  if (i < CDIM * CDIM) {
    oq[i] = (__bf16)wq[i];
    ok[i] = (__bf16)wk[i];
    ov[i] = (__bf16)wv[i];
    oo[i] = (__bf16)wo[i];
  }
}

// ---------- GroupNorm: one block per group (16 chans x 4096 = contiguous 65536 floats) ----------
__global__ __launch_bounds__(256) void gn_kernel(const float* __restrict__ xin,
                                                 const float* __restrict__ w,
                                                 const float* __restrict__ b,
                                                 float* __restrict__ out_f32,   // chan-major, may be null
                                                 __bf16* __restrict__ out_tok)  // token-major bf16
{
  const int GSZ = 16 * NTOK;
  int g = blockIdx.x;
  int tid = threadIdx.x;
  const float* xg = xin + (size_t)g * GSZ;
  float s = 0.f, ss = 0.f;
  for (int i = tid; i < GSZ; i += 256) { float v = xg[i]; s += v; ss += v * v; }
  __shared__ float rs[256], rss[256];
  rs[tid] = s; rss[tid] = ss;
  __syncthreads();
  for (int step = 128; step > 0; step >>= 1) {
    if (tid < step) { rs[tid] += rs[tid + step]; rss[tid] += rss[tid + step]; }
    __syncthreads();
  }
  float mean = rs[0] * (1.0f / GSZ);
  float var  = rss[0] * (1.0f / GSZ) - mean * mean;
  float rinv = rsqrtf(var + 1e-6f);
  for (int i = tid; i < GSZ; i += 256) {
    int c = g * 16 + (i >> 12);
    int n = i & (NTOK - 1);
    float v = (xg[i] - mean) * rinv * w[c] + b[c];
    if (out_f32) out_f32[(size_t)g * GSZ + i] = v;
    out_tok[(size_t)n * CDIM + c] = (__bf16)v;
  }
}

// ---------- bf16 WMMA GEMM: D[512 x 4096] = A[512 x 512] * Btok^T, + bias ----------
// mode 0: bf16 token-major out [N,O] (scaled by out_scale);
// mode 1: bf16 chan-major out [O,N] (scaled);
// mode 2: f32 chan-major out = (gemm + residual) * (1/sqrt(2)).
__global__ __launch_bounds__(256) void gemm_bf16_kernel(const __bf16* __restrict__ A,
                                                        const __bf16* __restrict__ Btok,
                                                        const float* __restrict__ bias,
                                                        void* __restrict__ outp,
                                                        const float* __restrict__ residual,
                                                        float out_scale, int mode) {
  const int O = CDIM, K = CDIM, N = NTOK;
  int wid  = blockIdx.x * 8 + (threadIdx.x >> 5);
  int mt   = wid & 31;          // 32 m-tiles of 16
  int n0   = (wid >> 5) * 64;   // 64 n-chunks of 64
  int lane = threadIdx.x & 31;
  int col = lane & 15, half = lane >> 4;

  v8f acc[4] = {v8f{}, v8f{}, v8f{}, v8f{}};
  for (int kk = 0; kk < K; kk += 32) {
    v16bf a = load_frag_a(A + (size_t)mt * 16 * K + kk, K);
#pragma unroll
    for (int t = 0; t < 4; ++t) {
      v16bf bfr = load_frag_b(Btok + (size_t)(n0 + t * 16) * K + kk, K);
      acc[t] = WMMA_BF16(a, bfr, acc[t]);
    }
  }
  float bv[8];
#pragma unroll
  for (int r = 0; r < 8; ++r) bv[r] = bias[mt * 16 + r + 8 * half];

#pragma unroll
  for (int t = 0; t < 4; ++t) {
#pragma unroll
    for (int r = 0; r < 8; ++r) {
      int m = mt * 16 + r + 8 * half;   // C/D layout: VGPR r -> row r + 8*half
      int n = n0 + t * 16 + col;        // lane%16 -> column
      float v = acc[t][r] + bv[r];
      if (mode == 0)      ((__bf16*)outp)[(size_t)n * O + m] = (__bf16)(v * out_scale);
      else if (mode == 1) ((__bf16*)outp)[(size_t)m * N + n] = (__bf16)(v * out_scale);
      else ((float*)outp)[(size_t)m * N + n] =
               (v + residual[(size_t)m * N + n]) * 0.70710678118654752f;
    }
  }
}

// ---------- flash attention ----------
// Block = 8 waves sharing one head; each wave owns a 16-query tile.
// K/V tiles (64 keys) staged in LDS via async-to-LDS, double-buffered.
// Softmax scale (1/sqrt(32)) is pre-folded into q.
__global__ __launch_bounds__(256) void attn_kernel(const __bf16* __restrict__ qtok,
                                                   const __bf16* __restrict__ ktok,
                                                   const __bf16* __restrict__ vchan,
                                                   __bf16* __restrict__ otok) {
  __shared__ __align__(16) __bf16 kbuf[2][64][32];   // [buf][key][dim]      8 KB
  __shared__ __align__(16) __bf16 vbuf[2][32][64];   // [buf][dim][key]      8 KB
  __shared__ __align__(16) __bf16 pt[8][16 * 64];    // per-wave P staging  16 KB

  int w    = threadIdx.x >> 5;
  int lane = threadIdx.x & 31;
  int col = lane & 15, half = lane >> 4;
  int h  = blockIdx.x;                 // head
  int qt = blockIdx.y * 8 + w;         // q tile (0..255)
  int t  = threadIdx.x;

  // cooperative 16B-per-thread copy slices (256 threads cover 4 KB per tile)
  int krow = t >> 2, kce = (t & 3) * 8;   // K tile: 64 rows x 32 dims
  int vrow = t >> 3, vce = (t & 7) * 8;   // V tile: 32 dims x 64 keys
  const __bf16* kg = ktok + (size_t)krow * CDIM + h * HDIM + kce;
  const __bf16* vg = vchan + (size_t)(h * HDIM + vrow) * NTOK + vce;

  v16bf qa = load_frag_a(qtok + (size_t)(qt * 16) * CDIM + h * HDIM, CDIM);
  v8f o0 = v8f{}, o1 = v8f{};
  float M[8], L[8];
#pragma unroll
  for (int r = 0; r < 8; ++r) { M[r] = -1e30f; L[r] = 0.f; }

  __bf16* myp = &pt[w][0];

  // preload chunk 0 into buffer 0
  async_copy_b128(kg, &kbuf[0][krow][kce]);
  async_copy_b128(vg, &vbuf[0][vrow][vce]);

  for (int j0 = 0; j0 < NTOK; j0 += 64) {
    int buf = (j0 >> 6) & 1;
    wait_async_copies();
    __syncthreads();  // all waves' copies landed; prior-iter LDS reads done
    if (j0 + 64 < NTOK) {
      async_copy_b128(kg + (size_t)(j0 + 64) * CDIM, &kbuf[buf ^ 1][krow][kce]);
      async_copy_b128(vg + (j0 + 64), &vbuf[buf ^ 1][vrow][vce]);
    }

    // S = q k^T for 64 keys (4 16x16 tiles), q pre-scaled
    v8f sc[4];
#pragma unroll
    for (int c = 0; c < 4; ++c) {
      v16bf kb = load_frag_b(&kbuf[buf][c * 16][0], 32);
      sc[c] = WMMA_BF16(qa, kb, v8f{});
    }

    // online softmax over the 64-key chunk
#pragma unroll
    for (int r = 0; r < 8; ++r) {
      float cm = fmaxf(fmaxf(sc[0][r], sc[1][r]), fmaxf(sc[2][r], sc[3][r]));
#pragma unroll
      for (int msk = 1; msk < 16; msk <<= 1) cm = fmaxf(cm, __shfl_xor(cm, msk, 32));
      float nm    = fmaxf(M[r], cm);
      float alpha = __expf(M[r] - nm);
      float p0 = __expf(sc[0][r] - nm), p1 = __expf(sc[1][r] - nm);
      float p2 = __expf(sc[2][r] - nm), p3 = __expf(sc[3][r] - nm);
      float rsum = (p0 + p1) + (p2 + p3);
#pragma unroll
      for (int msk = 1; msk < 16; msk <<= 1) rsum += __shfl_xor(rsum, msk, 32);
      L[r] = L[r] * alpha + rsum;
      M[r] = nm;
      o0[r] *= alpha;
      o1[r] *= alpha;
      int rr = (r + 8 * half) * 64 + col;   // D layout -> row-major 16x64
      myp[rr]      = (__bf16)p0;
      myp[rr + 16] = (__bf16)p1;
      myp[rr + 32] = (__bf16)p2;
      myp[rr + 48] = (__bf16)p3;
    }

    // O += P @ V (two K=32 sub-chunks x two 16-wide d tiles); same-wave LDS, no barrier
#pragma unroll
    for (int s = 0; s < 2; ++s) {
      v16bf pa = load_frag_a(myp + s * 32, 64);
      v16bf vb0 = load_frag_b(&vbuf[buf][0][s * 32], 64);
      v16bf vb1 = load_frag_b(&vbuf[buf][16][s * 32], 64);
      o0 = WMMA_BF16(pa, vb0, o0);
      o1 = WMMA_BF16(pa, vb1, o1);
    }
  }

#pragma unroll
  for (int r = 0; r < 8; ++r) {
    float inv = 1.f / L[r];
    int n = qt * 16 + r + 8 * half;
    otok[(size_t)n * CDIM + h * HDIM + col]      = (__bf16)(o0[r] * inv);
    otok[(size_t)n * CDIM + h * HDIM + 16 + col] = (__bf16)(o1[r] * inv);
  }
}

// ---------- host launch ----------
extern "C" void kernel_launch(void* const* d_in, const int* in_sizes, int n_in,
                              void* d_out, int out_size, void* d_ws, size_t ws_size,
                              hipStream_t stream) {
  const float* x   = (const float*)d_in[0];
  const float* gnw = (const float*)d_in[1];
  const float* gnb = (const float*)d_in[2];
  const float* wq  = (const float*)d_in[3];
  const float* bq  = (const float*)d_in[4];
  const float* wk  = (const float*)d_in[5];
  const float* bk  = (const float*)d_in[6];
  const float* wv  = (const float*)d_in[7];
  const float* bv  = (const float*)d_in[8];
  const float* wo  = (const float*)d_in[9];
  const float* bo  = (const float*)d_in[10];
  float* out = (float*)d_out;

  char* ws = (char*)d_ws;
  size_t off = 0;
  const size_t ACT_F32 = (size_t)CDIM * NTOK * 4;  // 8 MB
  const size_t ACT_BF  = (size_t)CDIM * NTOK * 2;  // 4 MB
  const size_t W_BF    = (size_t)CDIM * CDIM * 2;  // 0.5 MB

  float*  qn_f32 = (float*)(ws + off);  off += ACT_F32;
  __bf16* qn_tok = (__bf16*)(ws + off); off += ACT_BF;
  __bf16* vn_tok = (__bf16*)(ws + off); off += ACT_BF;
  __bf16* wqb    = (__bf16*)(ws + off); off += W_BF;
  __bf16* wkb    = (__bf16*)(ws + off); off += W_BF;
  __bf16* wvb    = (__bf16*)(ws + off); off += W_BF;
  __bf16* wob    = (__bf16*)(ws + off); off += W_BF;
  __bf16* q_tok  = (__bf16*)(ws + off); off += ACT_BF;
  __bf16* k_tok  = (__bf16*)(ws + off); off += ACT_BF;
  __bf16* v_chan = (__bf16*)(ws + off); off += ACT_BF;
  __bf16* a_tok  = (__bf16*)(ws + off); off += ACT_BF;

  const float kScale = 0.17677669529663688f;  // 1/sqrt(head_dim), folded into q

  cvt_w_kernel<<<(CDIM * CDIM + 255) / 256, 256, 0, stream>>>(wq, wk, wv, wo, wqb, wkb, wvb, wob);

  // qn = GN(x); vn = GN(qn) (reference applies the norm twice)
  gn_kernel<<<32, 256, 0, stream>>>(x, gnw, gnb, qn_f32, qn_tok);
  gn_kernel<<<32, 256, 0, stream>>>(qn_f32, gnw, gnb, nullptr, vn_tok);

  // q = (wq*qn + bq)*scale (token-major), k = wk*vn + bk (token-major), v = wv*vn + bv (chan-major)
  gemm_bf16_kernel<<<256, 256, 0, stream>>>(wqb, qn_tok, bq, q_tok, nullptr, kScale, 0);
  gemm_bf16_kernel<<<256, 256, 0, stream>>>(wkb, vn_tok, bk, k_tok, nullptr, 1.0f, 0);
  gemm_bf16_kernel<<<256, 256, 0, stream>>>(wvb, vn_tok, bv, v_chan, nullptr, 1.0f, 1);

  // flash attention: 16 heads x 256 q-tiles (8 waves/block, shared K/V LDS staging)
  attn_kernel<<<dim3(HEADS, 32), 256, 0, stream>>>(q_tok, k_tok, v_chan, a_tok);

  // out = (wo*attn + bo + x) / sqrt(2), fp32 chan-major
  gemm_bf16_kernel<<<256, 256, 0, stream>>>(wob, a_tok, bo, out, x, 1.0f, 2);
}
